// GraphSAGE_38946763440879
// MI455X (gfx1250) — matrix operand order repaired and checked by reading, not compiled, not run on hardware
//
#include <hip/hip_runtime.h>

#define D_DIM 768

typedef __attribute__((ext_vector_type(16))) __bf16 v16bf;
typedef __attribute__((ext_vector_type(8)))  float  v8f;

// ---------- small helper kernels ----------

__global__ void fill_f32_kernel(float* __restrict__ p, size_t n, float v) {
    size_t i = (size_t)blockIdx.x * blockDim.x + threadIdx.x;
    if (i < n) p[i] = v;
}

__global__ void cvt_f32_bf16_kernel(const float* __restrict__ in,
                                    __bf16* __restrict__ out, size_t n) {
    size_t i = (size_t)blockIdx.x * blockDim.x + threadIdx.x;
    if (i < n) out[i] = (__bf16)in[i];
}

// Scatter: one block per edge; atomically accumulate src features at dst.
// feat_f32 != nullptr -> read f32; else read bf16 from feat_b16.
__global__ void scatter_kernel(const float* __restrict__ feat_f32,
                               const __bf16* __restrict__ feat_b16,
                               const long long* __restrict__ ei, int E,
                               float* __restrict__ agg, float* __restrict__ deg,
                               int compute_deg) {
    int e = blockIdx.x;
    if (e >= E) return;
    int s = (int)ei[e];
    int d = (int)ei[(size_t)E + e];
    size_t sb = (size_t)s * D_DIM;
    size_t db = (size_t)d * D_DIM;
    for (int k = threadIdx.x; k < D_DIM; k += blockDim.x) {
        float v = feat_f32 ? feat_f32[sb + k] : (float)feat_b16[sb + k];
        atomicAdd(&agg[db + k], v);
    }
    if (compute_deg && threadIdx.x == 0) atomicAdd(&deg[d], 1.0f);
}

// agg_bf16[n,k] = bf16(agg_f32[n,k] / max(deg[n],1));  one block per node
__global__ void finalize_kernel(const float* __restrict__ aggf,
                                const float* __restrict__ deg,
                                __bf16* __restrict__ aggb, int Nn) {
    int n = blockIdx.x;
    if (n >= Nn) return;
    float inv = 1.0f / fmaxf(deg[n], 1.0f);
    size_t b = (size_t)n * D_DIM;
    for (int k = threadIdx.x; k < D_DIM; k += blockDim.x)
        aggb[b + k] = (__bf16)(aggf[b + k] * inv);
}

// ---------- fragment loaders ----------

// A fragment (16x32 bf16, MxK): lane<16 holds row M=lane, K = 0..15;
// lane>=16 holds K = 16..31. Caller passes p = row + k0 + half*16.
__device__ __forceinline__ v16bf load_a_frag(const __bf16* p) {
    union { uint4 u[2]; v16bf v; } t;
    t.u[0] = *(const uint4*)(p);        // 8 bf16: K base+0..7
    t.u[1] = *(const uint4*)(p + 8);    // 8 bf16: K base+8..15
    return t.v;
}

// B fragment (32x16 bf16, KxN): lane holds column N=lane%16, 16 contiguous K
// values starting at k0 + (lane>=16 ? 16 : 0). W stored [out,in] row-major, so
// a column of B == contiguous row slice of W. p = &W[j*D + k0 + half*16].
__device__ __forceinline__ v16bf load_b_frag(const __bf16* p) {
    union { uint4 u[2]; v16bf v; } t;
    t.u[0] = *(const uint4*)(p);
    t.u[1] = *(const uint4*)(p + 8);
    return t.v;
}

// ---------- fused dual-GEMM:  out = act( Aagg@Bl^T + bias + Aself@Br^T ) ----
// Block = 256 threads (8 waves), tile 128(M) x 128(N). Wave grid 4x2,
// each wave: 32x64 = 2x4 subtiles of 16x16. K = 768, stepped by 32.

__global__ __launch_bounds__(256) void sage_gemm_kernel(
    const __bf16* __restrict__ Aagg, const __bf16* __restrict__ Aself,
    const __bf16* __restrict__ Bl,   const __bf16* __restrict__ Br,
    const float* __restrict__ bias,
    float* __restrict__ outF, __bf16* __restrict__ outB,
    int M, int relu) {

    const int lane = threadIdx.x & 31;
    const int wave = threadIdx.x >> 5;
    const int wm = wave >> 1;            // 0..3
    const int wn = wave & 1;             // 0..1
    const int half = lane >> 4;          // 0/1
    const int l16 = lane & 15;
    const int rowBase = blockIdx.x * 128 + wm * 32;
    const int colBase = blockIdx.y * 128 + wn * 64;

    v8f acc[2][4];
#pragma unroll
    for (int mi = 0; mi < 2; ++mi)
#pragma unroll
        for (int ni = 0; ni < 4; ++ni) acc[mi][ni] = (v8f)(0.0f);

    for (int k0 = 0; k0 < D_DIM; k0 += 32) {
        v16bf aA[2], aS[2];
#pragma unroll
        for (int mi = 0; mi < 2; ++mi) {
            int r = rowBase + mi * 16 + l16;
            if (r >= M) r = M - 1;                      // clamp; store guarded
            size_t off = (size_t)r * D_DIM + k0 + half * 16;
            aA[mi] = load_a_frag(Aagg + off);
            aS[mi] = load_a_frag(Aself + off);
        }
        v16bf bL[4], bR[4];
#pragma unroll
        for (int ni = 0; ni < 4; ++ni) {
            int j = colBase + ni * 16 + l16;
            size_t off = (size_t)j * D_DIM + k0 + half * 16;
            bL[ni] = load_b_frag(Bl + off);
            bR[ni] = load_b_frag(Br + off);
        }
#pragma unroll
        for (int mi = 0; mi < 2; ++mi)
#pragma unroll
            for (int ni = 0; ni < 4; ++ni) {
                acc[mi][ni] = __builtin_amdgcn_wmma_f32_16x16x32_bf16(
                    false, aA[mi], false, bL[ni], (short)0, acc[mi][ni],
                    false, false);
                acc[mi][ni] = __builtin_amdgcn_wmma_f32_16x16x32_bf16(
                    false, aS[mi], false, bR[ni], (short)0, acc[mi][ni],
                    false, false);
            }
    }

    // Epilogue: C/D layout — VGPR v: row = base + v + (lane>=16 ? 8 : 0),
    // col = base + lane%16.
#pragma unroll
    for (int mi = 0; mi < 2; ++mi)
#pragma unroll
        for (int ni = 0; ni < 4; ++ni) {
            int col = colBase + ni * 16 + l16;
            float bv = bias[col];
#pragma unroll
            for (int v = 0; v < 8; ++v) {
                int row = rowBase + mi * 16 + half * 8 + v;
                if (row < M) {
                    float val = acc[mi][ni][v] + bv;
                    if (relu) val = fmaxf(val, 0.0f);
                    size_t o = (size_t)row * D_DIM + col;
                    if (outF) outF[o] = val;
                    if (outB) outB[o] = (__bf16)val;
                }
            }
        }
}

// ---------- launcher ----------

static inline size_t align256(size_t v) { return (v + 255) & ~(size_t)255; }

extern "C" void kernel_launch(void* const* d_in, const int* in_sizes, int n_in,
                              void* d_out, int out_size, void* d_ws,
                              size_t ws_size, hipStream_t stream) {
    const float*     x   = (const float*)d_in[0];
    const long long* ei  = (const long long*)d_in[1];
    const float*     W1l = (const float*)d_in[2];
    const float*     b1l = (const float*)d_in[3];
    const float*     W1r = (const float*)d_in[4];
    const float*     W2l = (const float*)d_in[5];
    const float*     b2l = (const float*)d_in[6];
    const float*     W2r = (const float*)d_in[7];

    const int N  = in_sizes[0] / D_DIM;          // 50000
    const int E  = in_sizes[1] / 2;              // 100000
    const size_t ND = (size_t)N * D_DIM;
    const size_t DD = (size_t)D_DIM * D_DIM;

    float* out = (float*)d_out;                  // also reused as f32 agg scratch

    char* ws = (char*)d_ws;
    size_t off = 0;
    float*  deg  = (float*)(ws + off);  off = align256(off + (size_t)N * 4);
    __bf16* xb   = (__bf16*)(ws + off); off = align256(off + ND * 2);
    __bf16* aggb = (__bf16*)(ws + off); off = align256(off + ND * 2);
    __bf16* hb   = (__bf16*)(ws + off); off = align256(off + ND * 2);
    __bf16* w1l  = (__bf16*)(ws + off); off = align256(off + DD * 2);
    __bf16* w1r  = (__bf16*)(ws + off); off = align256(off + DD * 2);
    __bf16* w2l  = (__bf16*)(ws + off); off = align256(off + DD * 2);
    __bf16* w2r  = (__bf16*)(ws + off); off = align256(off + DD * 2);

    const int T = 256;
    dim3 gemm_grid((N + 127) / 128, D_DIM / 128);

    // Precision prep: bf16 copies of activations + weights (weights keep
    // [out,in] row-major — that IS the WMMA B layout, no transpose needed).
    cvt_f32_bf16_kernel<<<(int)((ND + T - 1) / T), T, 0, stream>>>(x, xb, ND);
    cvt_f32_bf16_kernel<<<(int)((DD + T - 1) / T), T, 0, stream>>>(W1l, w1l, DD);
    cvt_f32_bf16_kernel<<<(int)((DD + T - 1) / T), T, 0, stream>>>(W1r, w1r, DD);
    cvt_f32_bf16_kernel<<<(int)((DD + T - 1) / T), T, 0, stream>>>(W2l, w2l, DD);
    cvt_f32_bf16_kernel<<<(int)((DD + T - 1) / T), T, 0, stream>>>(W2r, w2r, DD);

    // ---- layer 1 aggregation (f32 accumulate in d_out scratch) ----
    fill_f32_kernel<<<(int)((ND + T - 1) / T), T, 0, stream>>>(out, ND, 0.0f);
    fill_f32_kernel<<<(N + T - 1) / T, T, 0, stream>>>(deg, (size_t)N, 0.0f);
    scatter_kernel<<<E, T, 0, stream>>>(x, nullptr, ei, E, out, deg, 1);
    finalize_kernel<<<N, T, 0, stream>>>(out, deg, aggb, N);

    // ---- layer 1 fused dual-GEMM + bias + ReLU -> h (bf16) ----
    sage_gemm_kernel<<<gemm_grid, T, 0, stream>>>(aggb, xb, w1l, w1r, b1l,
                                                  nullptr, hb, N, 1);

    // ---- layer 2 aggregation ----
    fill_f32_kernel<<<(int)((ND + T - 1) / T), T, 0, stream>>>(out, ND, 0.0f);
    scatter_kernel<<<E, T, 0, stream>>>(nullptr, hb, ei, E, out, deg, 0);
    finalize_kernel<<<N, T, 0, stream>>>(out, deg, aggb, N);

    // ---- layer 2 fused dual-GEMM + bias -> d_out (f32, fully overwritten) --
    sage_gemm_kernel<<<gemm_grid, T, 0, stream>>>(aggb, hb, w2l, w2r, b2l,
                                                  out, nullptr, N, 0);
}
